// SparsePoolingLayer_15934328669015
// MI455X (gfx1250) — matrix operand order repaired and checked by reading, not compiled, not run on hardware
//
#include <hip/hip_runtime.h>
#include <hip/hip_bf16.h>
#include <math.h>

typedef _Float16 v8h  __attribute__((ext_vector_type(8)));
typedef _Float16 v16h __attribute__((ext_vector_type(16)));
typedef float    v8f  __attribute__((ext_vector_type(8)));
typedef float    v4f  __attribute__((ext_vector_type(4)));

#define TAU 0.5f
#define EPSF 0.001f

// Problem constants (from setup_inputs)
constexpr int B_   = 32;
constexpr int CIN  = 64;
constexpr int HIN  = 64;
constexpr int WIN  = 64;
constexpr int COUT = 256;
constexpr int KS   = 5;
constexpr int HO   = 60;
constexpr int WO   = 60;
constexpr int SPAT = HO * WO;                 // 3600
constexpr size_t NTOT = (size_t)B_ * COUT * SPAT; // 29,491,200 elements

__device__ __forceinline__ int laneid() { return threadIdx.x & 31; }

// A fragment (16x32 f16, M x K), A row-major, lda elements, K contiguous.
// ISA layout: lanes 0-15 hold M=lane, K {0..7, 16..23}; lanes 16-31 hold M=lane-16, K {8..15, 24..31}.
__device__ __forceinline__ v16h load_frag_A(const _Float16* __restrict__ A, int lda, int m0, int k0) {
    int l = laneid();
    const _Float16* p = A + (size_t)(m0 + (l & 15)) * lda + (size_t)(k0 + ((l >> 4) << 3));
    v8h lo = *(const v8h*)p;          // K: base..base+7
    v8h hi = *(const v8h*)(p + 16);   // K: base+16..base+23
    return __builtin_shufflevector(lo, hi, 0,1,2,3,4,5,6,7,8,9,10,11,12,13,14,15);
}

// B fragment (32x16 f16, K x N). Storage is column-of-K contiguous: element (k, col) at colp[k].
// colp must already be the per-lane column pointer (col = n0 + (lane&15)).
// ISA layout: lanes 0-15 hold col, K {0..7, 16..23}; lanes 16-31 hold col, K {8..15, 24..31}.
__device__ __forceinline__ v16h load_frag_B(const _Float16* __restrict__ colp, int k0) {
    int l = laneid();
    const _Float16* p = colp + k0 + ((l >> 4) << 3);
    v8h lo = *(const v8h*)p;          // K: base..base+7
    v8h hi = *(const v8h*)(p + 16);   // K: base+16..base+23
    return __builtin_shufflevector(lo, hi, 0,1,2,3,4,5,6,7,8,9,10,11,12,13,14,15);
}

__device__ __forceinline__ v8f wmma16(v16h a, v16h b, v8f c) {
    return __builtin_amdgcn_wmma_f32_16x16x32_f16(false, a, false, b, (short)0, c, false, false);
}

// ---------------- conversion / packing kernels (run once per call) ----------------

// W_rec (256x256 f32 row-major [o][i]) -> f16 same layout. Thread 0 also re-inits control block.
__global__ void cvt_wrec(const float* __restrict__ Wrec, _Float16* __restrict__ out,
                         int* __restrict__ flag, float* __restrict__ sums) {
    int i = blockIdx.x * blockDim.x + threadIdx.x;
    if (i == 0) { *flag = 0; sums[0] = 0.f; sums[1] = 0.f; }
    if (i < COUT * COUT) out[i] = (_Float16)Wrec[i];
}

// W_ff (256,64,5,5) -> packed f16 [ky*5+kx][o][i] (i contiguous)
__global__ void cvt_wff(const float* __restrict__ Wff, _Float16* __restrict__ out) {
    int t = blockIdx.x * blockDim.x + threadIdx.x;   // per (kykx, o): 25*256 = 6400
    if (t >= 25 * COUT) return;
    int o  = t & 255;
    int kk = t >> 8;                                  // ky*5+kx
    const float* src = Wff + (size_t)o * (CIN * KS * KS) + kk;  // stride 25 along i
    _Float16* dst = out + (size_t)t * CIN;
    #pragma unroll
    for (int i = 0; i < CIN; ++i) dst[i] = (_Float16)src[(size_t)i * (KS * KS)];
}

// x (32,64,64,64 NCHW) -> f16 [b][h][w][i] (i contiguous)
__global__ void cvt_x(const float* __restrict__ x, _Float16* __restrict__ out) {
    int t = blockIdx.x * blockDim.x + threadIdx.x;   // per (b,h,w): 131072
    if (t >= B_ * HIN * WIN) return;
    int w = t & 63, h = (t >> 6) & 63, b = t >> 12;
    const float* src = x + (size_t)b * CIN * HIN * WIN + (size_t)h * WIN + w; // stride 4096 along i
    _Float16* dst = out + (size_t)t * CIN;
    #pragma unroll
    for (int i = 0; i < CIN; ++i) dst[i] = (_Float16)src[(size_t)i * (HIN * WIN)];
}

// ---------------- conv: u0 = conv2d(x, W_ff); u = 0; a0 = relu(u0 - thr) ----------------
// 25 shifted K=64 GEMMs via WMMA. Internal layouts: u0,u fp32 [b][s][o]; a f16 [b][s][o].
__global__ __launch_bounds__(256) void conv_init(const _Float16* __restrict__ xh,
                                                 const _Float16* __restrict__ wffp,
                                                 const float* __restrict__ thr,
                                                 float* __restrict__ u0,
                                                 float* __restrict__ u,
                                                 _Float16* __restrict__ a0) {
    int wv  = threadIdx.x >> 5;
    int lin = blockIdx.x * 8 + wv;      // 16 m-tiles * 60 rows * 32 batches = 30720 waves
    int mt   = lin & 15;
    int rest = lin >> 4;
    int y = rest % HO;
    int b = rest / HO;
    int m0 = mt * 16;
    int l = laneid();
    int c = l & 15;

    v8f acc[4];
    #pragma unroll
    for (int j = 0; j < 4; ++j) acc[j] = v8f{0.f,0.f,0.f,0.f,0.f,0.f,0.f,0.f};

    for (int ky = 0; ky < KS; ++ky) {
        const _Float16* xrow = xh + (((size_t)b * HIN + (y + ky)) * WIN) * CIN;
        #pragma unroll
        for (int kx = 0; kx < KS; ++kx) {
            const _Float16* ap = wffp + (size_t)(ky * KS + kx) * COUT * CIN;  // [o][i], lda=64
            // issue all fragment loads of this (ky,kx) step as one clause, then 8 WMMAs
            v16h af0 = load_frag_A(ap, CIN, m0, 0);
            v16h af1 = load_frag_A(ap, CIN, m0, 32);
            v16h bf[8];
            #pragma unroll
            for (int j = 0; j < 4; ++j) {
                int xin = j * 16 + c + kx;
                if (xin > WIN - 1) xin = WIN - 1;      // clamp (masked on store)
                const _Float16* colp = xrow + (size_t)xin * CIN;
                bf[2 * j]     = load_frag_B(colp, 0);
                bf[2 * j + 1] = load_frag_B(colp, 32);
            }
            #pragma unroll
            for (int j = 0; j < 4; ++j) {
                acc[j] = wmma16(af0, bf[2 * j], acc[j]);
                acc[j] = wmma16(af1, bf[2 * j + 1], acc[j]);
            }
        }
    }

    int mrow = m0 + ((l >> 4) << 3);
    #pragma unroll
    for (int j = 0; j < 4; ++j) {
        int n = j * 16 + c;
        if (n < WO) {
            int s = y * WO + n;
            size_t idx = ((size_t)b * SPAT + s) * COUT + mrow;
            float z[8]; _Float16 ah[8];
            #pragma unroll
            for (int r = 0; r < 8; ++r) {
                z[r] = acc[j][r];
                float av = z[r] - thr[mrow + r];
                ah[r] = (_Float16)(av > 0.f ? av : 0.f);
            }
            *(v4f*)(u0 + idx)     = v4f{z[0], z[1], z[2], z[3]};
            *(v4f*)(u0 + idx + 4) = v4f{z[4], z[5], z[6], z[7]};
            *(v4f*)(u  + idx)     = v4f{0.f, 0.f, 0.f, 0.f};
            *(v4f*)(u  + idx + 4) = v4f{0.f, 0.f, 0.f, 0.f};
            *(v8h*)(a0 + idx)     = v8h{ah[0], ah[1], ah[2], ah[3], ah[4], ah[5], ah[6], ah[7]};
        }
    }
}

// ---------------- one fixed-point iteration: rec = W_rec @ a; u,a update; norm accumulation ----------------
__global__ __launch_bounds__(256) void lca_update(const int* __restrict__ flag,
                                                  float* __restrict__ sums,
                                                  const _Float16* __restrict__ Wrec,
                                                  const _Float16* __restrict__ a_in,
                                                  const float* __restrict__ u0,
                                                  float* __restrict__ u,
                                                  _Float16* __restrict__ a_new,
                                                  const float* __restrict__ thr) {
    if (*flag) return;                   // converged / iteration-bound reached: frozen

    int wv  = threadIdx.x >> 5;
    int lin = blockIdx.x * 8 + wv;       // 4 m-groups * 225 s-tiles * 32 batches = 28800 waves
    int mg   = lin & 3;
    int rest = lin >> 2;
    int t = rest % (SPAT / 16);          // 225
    int b = rest / (SPAT / 16);
    int s0 = t * 16;
    int m0 = mg * 64;
    int l = laneid();
    int c = l & 15;
    int mrow_off = (l >> 4) << 3;

    const _Float16* colp = a_in + ((size_t)b * SPAT + s0 + c) * COUT;

    // Prefetch the epilogue operands (u, u0) so their lines arrive during the GEMM.
    size_t ebase = ((size_t)b * SPAT + s0 + c) * COUT;
    #pragma unroll
    for (int j = 0; j < 4; ++j) {
        size_t idx = ebase + m0 + j * 16 + mrow_off;
        __builtin_prefetch(u + idx, 0, 1);
        __builtin_prefetch(u0 + idx, 0, 1);
    }

    v8f acc[4];
    #pragma unroll
    for (int j = 0; j < 4; ++j) acc[j] = v8f{0.f,0.f,0.f,0.f,0.f,0.f,0.f,0.f};

    // Software-pipelined K loop: next step's 1 B + 4 A fragments load while current 4 WMMAs run.
    v16h bf = load_frag_B(colp, 0);
    v16h af[4];
    #pragma unroll
    for (int j = 0; j < 4; ++j) af[j] = load_frag_A(Wrec, COUT, m0 + j * 16, 0);

    #pragma unroll
    for (int kk = 0; kk < COUT; kk += 32) {
        v16h bf_n = bf;
        v16h af_n[4] = {af[0], af[1], af[2], af[3]};
        if (kk + 32 < COUT) {
            bf_n = load_frag_B(colp, kk + 32);
            #pragma unroll
            for (int j = 0; j < 4; ++j)
                af_n[j] = load_frag_A(Wrec, COUT, m0 + j * 16, kk + 32);
        }
        #pragma unroll
        for (int j = 0; j < 4; ++j) acc[j] = wmma16(af[j], bf, acc[j]);
        bf = bf_n;
        #pragma unroll
        for (int j = 0; j < 4; ++j) af[j] = af_n[j];
    }

    float d2 = 0.f, n2 = 0.f;
    #pragma unroll
    for (int j = 0; j < 4; ++j) {
        int m = m0 + j * 16 + mrow_off;
        size_t idx = ebase + m;
        v4f uo0 = *(const v4f*)(u + idx);
        v4f uo1 = *(const v4f*)(u + idx + 4);
        v4f z0  = *(const v4f*)(u0 + idx);
        v4f z1  = *(const v4f*)(u0 + idx + 4);
        float un[8]; _Float16 ah[8];
        #pragma unroll
        for (int r = 0; r < 8; ++r) {
            float uold = (r < 4) ? uo0[r] : uo1[r - 4];
            float zz   = (r < 4) ? z0[r]  : z1[r - 4];
            float v = (1.f - TAU) * uold + TAU * (zz - acc[j][r]);
            float d = v - uold;
            d2 += d * d;
            n2 += v * v;
            float av = v - thr[m + r];
            ah[r] = (_Float16)(av > 0.f ? av : 0.f);
            un[r] = v;
        }
        *(v4f*)(u + idx)     = v4f{un[0], un[1], un[2], un[3]};
        *(v4f*)(u + idx + 4) = v4f{un[4], un[5], un[6], un[7]};
        *(v8h*)(a_new + idx) = v8h{ah[0], ah[1], ah[2], ah[3], ah[4], ah[5], ah[6], ah[7]};
    }

    #pragma unroll
    for (int off = 16; off; off >>= 1) {
        d2 += __shfl_xor(d2, off, 32);
        n2 += __shfl_xor(n2, off, 32);
    }
    if (l == 0) {
        atomicAdd(&sums[0], d2);
        atomicAdd(&sums[1], n2);
    }
}

// Convergence / iteration-bound check (1 thread), zeroes accumulators for next iteration.
__global__ void lca_check(int* __restrict__ flag, float* __restrict__ sums,
                          const int* __restrict__ max_n_iter, int i_done) {
    if (!*flag) {
        float d2 = sums[0], n2 = sums[1];
        if (sqrtf(d2) < EPSF * sqrtf(n2)) *flag = 1;       // converged
        else if (i_done > 2 * (*max_n_iter)) *flag = 1;    // i <= max_iter bound
    }
    sums[0] = 0.f;
    sums[1] = 0.f;
}

// Final export: a_final = relu(u - thr), internal [b][s][o] fp32 -> NCHW fp32 d_out.
// LDS-tiled 64x64 transpose: coalesced reads along o, coalesced writes along s.
__global__ __launch_bounds__(256) void export_a(const float* __restrict__ u,
                                                const float* __restrict__ thr,
                                                float* __restrict__ out) {
    __shared__ float tile[64][65];                // padded: bank-conflict-free transpose
    int bid  = blockIdx.x;                        // 32 b * 57 s-tiles * 4 o-tiles
    int ot   = bid & 3;
    int rest = bid >> 2;
    int st   = rest % 57;
    int b    = rest / 57;
    int s0 = st * 64, o0 = ot * 64;
    int tl = threadIdx.x & 63;                    // fast dim
    int tr = threadIdx.x >> 6;                    // 0..3

    #pragma unroll
    for (int i = 0; i < 16; ++i) {
        int sl = tr + 4 * i;
        int s = s0 + sl;
        if (s < SPAT)
            tile[sl][tl] = u[((size_t)b * SPAT + s) * COUT + o0 + tl];
    }
    __syncthreads();
    #pragma unroll
    for (int i = 0; i < 16; ++i) {
        int ol = tr + 4 * i;
        int s = s0 + tl;
        if (s < SPAT) {
            float v = tile[tl][ol] - thr[o0 + ol];
            out[((size_t)b * COUT + o0 + ol) * SPAT + s] = v > 0.f ? v : 0.f;
        }
    }
}

extern "C" void kernel_launch(void* const* d_in, const int* in_sizes, int n_in,
                              void* d_out, int out_size, void* d_ws, size_t ws_size,
                              hipStream_t stream) {
    (void)in_sizes; (void)n_in; (void)out_size; (void)ws_size;
    const float* x     = (const float*)d_in[0];
    const float* Wff   = (const float*)d_in[1];
    const float* Wrec  = (const float*)d_in[2];
    const float* thr   = (const float*)d_in[3];
    const int*   maxit = (const int*)d_in[4];

    char* ws = (char*)d_ws;
    size_t off = 0;
    auto take = [&](size_t bytes) -> char* {
        char* p = ws + off;
        off += (bytes + 255) & ~(size_t)255;
        return p;
    };

    char*     ctl  = take(256);
    int*      flag = (int*)ctl;
    float*    sums = (float*)(ctl + 16);
    float*    u0   = (float*)take(NTOT * sizeof(float));
    float*    u    = (float*)take(NTOT * sizeof(float));
    _Float16* abuf0 = (_Float16*)take(NTOT * sizeof(_Float16));
    _Float16* abuf1 = (_Float16*)take(NTOT * sizeof(_Float16));
    _Float16* xh    = (_Float16*)take((size_t)B_ * HIN * WIN * CIN * sizeof(_Float16));
    _Float16* wrh   = (_Float16*)take((size_t)COUT * COUT * sizeof(_Float16));
    _Float16* wfp   = (_Float16*)take((size_t)KS * KS * COUT * CIN * sizeof(_Float16));

    // Pack / convert inputs; re-init control block every call.
    cvt_wrec<<<(COUT * COUT + 255) / 256, 256, 0, stream>>>(Wrec, wrh, flag, sums);
    cvt_wff<<<(25 * COUT + 255) / 256, 256, 0, stream>>>(Wff, wfp);
    cvt_x<<<(B_ * HIN * WIN + 255) / 256, 256, 0, stream>>>(x, xh);

    // u0 + u=0 + a0
    conv_init<<<(16 * HO * B_) / 8, 256, 0, stream>>>(xh, wfp, thr, u0, u, abuf0);

    // Fixed 41 launches (= max body executions for max_n_iter=20); device flag freezes the rest.
    const int UPD_BLOCKS = (4 * (SPAT / 16) * B_) / 8;   // 3600
    for (int i = 0; i < 41; ++i) {
        const _Float16* ain = (i & 1) ? abuf1 : abuf0;
        _Float16*      aout = (i & 1) ? abuf0 : abuf1;
        lca_update<<<UPD_BLOCKS, 256, 0, stream>>>(flag, sums, wrh, ain, u0, u, aout, thr);
        lca_check<<<1, 1, 0, stream>>>(flag, sums, maxit, i + 1);
    }

    // Coalesced transposing export of a_final.
    export_a<<<32 * 57 * 4, 256, 0, stream>>>(u, thr, (float*)d_out);
}
